// GLSTM_57277683859535
// MI455X (gfx1250) — compile-verified, matrix operand order checked by reading, not run on hardware
//
#include <hip/hip_runtime.h>

typedef unsigned short u16;
typedef __attribute__((ext_vector_type(16))) __bf16 v16bf;
typedef __attribute__((ext_vector_type(8)))  float  v8f;
typedef int v4i_t __attribute__((vector_size(16)));

#define T_STEPS 32
#define NNODES  2000
#define FIN     64
#define HDIM    512
#define GDIM    2048   /* 4*H */
#define EDGES   64000
#define LAYERS  2

#if __has_builtin(__builtin_amdgcn_global_load_async_to_lds_b128)
#define USE_ASYNC 1
#else
#define USE_ASYNC 0
#endif

#if USE_ASYNC
#define ASYNC_CP(ldst, gsrc)                                                 \
  __builtin_amdgcn_global_load_async_to_lds_b128(                            \
      (__attribute__((address_space(1))) v4i_t*)(gsrc),                      \
      (__attribute__((address_space(3))) v4i_t*)(ldst), 0, 0)
#else
#define ASYNC_CP(ldst, gsrc) (*(uint4*)(ldst) = *(const uint4*)(gsrc))
#endif

__device__ __forceinline__ void wait_async0() {
#if USE_ASYNC
#if __has_builtin(__builtin_amdgcn_s_wait_asynccnt)
  __builtin_amdgcn_s_wait_asynccnt(0);
#else
  asm volatile("s_wait_asynccnt 0x0" ::: "memory");
#endif
#endif
}

__device__ __forceinline__ u16 f2bf(float f) {
  unsigned u = __float_as_uint(f);
  unsigned r = u + 0x7FFFu + ((u >> 16) & 1u);   // round-to-nearest-even
  return (u16)(r >> 16);
}

union FragU { uint4 q[2]; v16bf v; };
union Cvt8 { u16 h[8]; uint4 q; };

// ---------------------------------------------------------------------------
// Generic bf16 WMMA GEMM: C(MxNout) = A(MxK) * Bt(NoutxK)^T  (+ epilogue)
// Double-buffered LDS, async global->LDS copies, 1 barrier / K-iteration.
// AMODE 0: A0 bf16, lda=K
// AMODE 1: Af fp32 (converted on load), lda=K
// AMODE 2: A = [A0 (k<512) | A1 (k>=512)], both bf16 lda=512, K=1024
// EPI   0: v=acc+bias; outB=bf16(v); rows<carryM also init h/c carries
// EPI   1: outF = acc + bias
// EPI   2: outF = acc + pre[row][col]
// ---------------------------------------------------------------------------
template<int AMODE, int EPI>
__global__ __launch_bounds__(256)
void gemm_wmma(const u16* __restrict__ A0, const u16* __restrict__ A1,
               const float* __restrict__ Af, const u16* __restrict__ Bt,
               int M, int K, int Nout,
               const float* __restrict__ bias, const float* __restrict__ pre,
               float* __restrict__ outF, u16* __restrict__ outB,
               u16* __restrict__ hbf, float* __restrict__ hf,
               float* __restrict__ cf, int carryM)
{
  // 128x32 bf16 tiles, row stride 40 elements (80 B, keeps b128 alignment,
  // spreads rows across banks), double buffered.
  __shared__ __align__(16) u16 As[2][128 * 40];
  __shared__ __align__(16) u16 Bs[2][128 * 40];

  const int tid  = threadIdx.x;
  const int lane = tid & 31;
  const int wave = tid >> 5;
  const int wm   = wave >> 2;        // 2 wave rows  (64 M each)
  const int wn   = wave & 3;         // 4 wave cols  (32 N each)
  const int tileM = blockIdx.y * 128;
  const int tileN = blockIdx.x * 128;
  const int mhalf = lane >> 4;       // K-half select per ISA 16-bit A layout
  const int mrow  = lane & 15;

  v8f acc[4][2];
#pragma unroll
  for (int i = 0; i < 4; ++i)
#pragma unroll
    for (int j = 0; j < 2; ++j)
      acc[i][j] = (v8f){0.f,0.f,0.f,0.f,0.f,0.f,0.f,0.f};

  // stage one 128x32 A tile + 128x32 B^T tile into LDS buffer `buf`
  auto stage = [&](int k0, int buf) {
#pragma unroll
    for (int it = 0; it < 2; ++it) {
      int s   = tid + it * 256;          // 0..511
      int row = s >> 2;
      int kc  = (s & 3) << 3;            // 0,8,16,24
      // ---- A ----
      {
        int grow = tileM + row; if (grow > M - 1) grow = M - 1;
        if (AMODE == 1) {
          const float* p = Af + (long)grow * K + k0 + kc;
          float4 f0 = *(const float4*)(p);
          float4 f1 = *(const float4*)(p + 4);
          Cvt8 c;
          c.h[0]=f2bf(f0.x); c.h[1]=f2bf(f0.y); c.h[2]=f2bf(f0.z); c.h[3]=f2bf(f0.w);
          c.h[4]=f2bf(f1.x); c.h[5]=f2bf(f1.y); c.h[6]=f2bf(f1.z); c.h[7]=f2bf(f1.w);
          *(uint4*)&As[buf][row * 40 + kc] = c.q;
        } else {
          const u16* p;
          if (AMODE == 2) {
            int kg = k0 + kc;
            p = (kg < 512) ? (A0 + (long)grow * 512 + kg)
                           : (A1 + (long)grow * 512 + (kg - 512));
          } else {
            p = A0 + (long)grow * K + k0 + kc;
          }
          if (k0 + 64 < K) __builtin_prefetch(p + 64, 0, 0);
          ASYNC_CP(&As[buf][row * 40 + kc], p);
        }
      }
      // ---- B ----
      {
        int gcol = tileN + row;
        const u16* p = Bt + (long)gcol * K + k0 + kc;
        if (k0 + 64 < K) __builtin_prefetch(p + 64, 0, 0);
        ASYNC_CP(&Bs[buf][row * 40 + kc], p);
      }
    }
  };

  const int nk = K >> 5;
  stage(0, 0);
  for (int ki = 0; ki < nk; ++ki) {
    wait_async0();        // my async copies for buffer (ki&1) are done
    __syncthreads();      // everyone's are done; prev buffer free for refill
    if (ki + 1 < nk) stage(ki + 1, (ki + 1) & 1);

    const u16* Asb = &As[ki & 1][0];
    const u16* Bsb = &Bs[ki & 1][0];
    FragU a[4], b[2];
#pragma unroll
    for (int im = 0; im < 4; ++im) {
      int r = wm * 64 + im * 16 + mrow;
      int base = r * 40 + mhalf * 8;
      a[im].q[0] = *(const uint4*)&Asb[base];
      a[im].q[1] = *(const uint4*)&Asb[base + 16];
    }
#pragma unroll
    for (int in = 0; in < 2; ++in) {
      int r = wn * 32 + in * 16 + mrow;
      int base = r * 40 + mhalf * 8;
      b[in].q[0] = *(const uint4*)&Bsb[base];
      b[in].q[1] = *(const uint4*)&Bsb[base + 16];
    }
#pragma unroll
    for (int im = 0; im < 4; ++im)
#pragma unroll
      for (int in = 0; in < 2; ++in)
        acc[im][in] = __builtin_amdgcn_wmma_f32_16x16x32_bf16(
            false, a[im].v, false, b[in].v, (short)0, acc[im][in], false, false);
  }

  // ---- epilogue ----
#pragma unroll
  for (int in = 0; in < 2; ++in) {
    int col = tileN + wn * 32 + in * 16 + mrow;
    float bv = (EPI == 2) ? 0.f : bias[col];
#pragma unroll
    for (int im = 0; im < 4; ++im) {
      int rbase = tileM + wm * 64 + im * 16 + mhalf * 8;
#pragma unroll
      for (int r = 0; r < 8; ++r) {
        int row = rbase + r;
        if (row < M) {
          float v = acc[im][in][r];
          if (EPI == 0) {
            v += bv;
            outB[(long)row * Nout + col] = f2bf(v);
            if (row < carryM) {
              hbf[(long)row * Nout + col] = f2bf(v);
              hf [(long)row * Nout + col] = v;
              cf [(long)row * Nout + col] = v;
            }
          } else if (EPI == 1) {
            outF[(long)row * Nout + col] = v + bv;
          } else {
            outF[(long)row * Nout + col] = v + pre[(long)row * Nout + col];
          }
        }
      }
    }
  }
}

// ---------------------------------------------------------------------------
// fp32 -> bf16 transpose: out[n*ostride + koff + k] = bf16(in[k*N + n])
// ---------------------------------------------------------------------------
__global__ __launch_bounds__(256)
void transpose_cvt(u16* __restrict__ out, const float* __restrict__ in,
                   int K, int N, int ostride, int koff)
{
  long gid = (long)blockIdx.x * 256 + threadIdx.x;
  long total = (long)K * N;
  if (gid >= total) return;
  long k = gid / N;
  long n = gid - k * N;
  out[n * ostride + koff + k] = f2bf(in[gid]);
}

__global__ __launch_bounds__(256)
void zero_u16(u16* __restrict__ p, long nq)   // nq in uint4 units
{
  long gid = (long)blockIdx.x * 256 + threadIdx.x;
  if (gid < nq) ((uint4*)p)[gid] = (uint4){0u,0u,0u,0u};
}

// ---------------------------------------------------------------------------
// Edge scatter: agg[dst] += h[src] with packed-bf16 global atomics.
// One thread handles one (edge, 8-element chunk): 1x b128 load + 4 pk atomics.
// ---------------------------------------------------------------------------
__global__ __launch_bounds__(256)
void scatter_pk(const long long* __restrict__ eidx, long srcBase, long dstBase,
                const u16* __restrict__ hbf, u16* __restrict__ agg)
{
  long gid = (long)blockIdx.x * 256 + threadIdx.x;
  int e  = (int)(gid >> 6);
  int k0 = (int)(gid & 63) << 3;
  int src = (int)eidx[srcBase + e];
  int dst = (int)eidx[dstBase + e];
  uint4 d = *(const uint4*)(hbf + (long)src * HDIM + k0);
  unsigned long long base = (unsigned long long)(agg + (long)dst * HDIM + k0);
  asm volatile("global_atomic_pk_add_bf16 %0, %1, off" :: "v"(base),         "v"(d.x) : "memory");
  asm volatile("global_atomic_pk_add_bf16 %0, %1, off" :: "v"(base + 4ull),  "v"(d.y) : "memory");
  asm volatile("global_atomic_pk_add_bf16 %0, %1, off" :: "v"(base + 8ull),  "v"(d.z) : "memory");
  asm volatile("global_atomic_pk_add_bf16 %0, %1, off" :: "v"(base + 12ull), "v"(d.w) : "memory");
}

// ---------------------------------------------------------------------------
// Elementwise LSTM cell: reads gate strips i|f|g|o, updates c, writes h (f32+bf16)
// ---------------------------------------------------------------------------
__global__ __launch_bounds__(256)
void lstm_cell(const float* __restrict__ gates, float* __restrict__ cf,
               float* __restrict__ hf, u16* __restrict__ hbf)
{
  long gid = (long)blockIdx.x * 256 + threadIdx.x;   // < N*H
  int m = (int)(gid >> 9);
  int k = (int)(gid & 511);
  const float* g0 = gates + (long)m * GDIM;
  float iv = g0[k], fv = g0[512 + k], gv = g0[1024 + k], ov = g0[1536 + k];
  iv = 1.f / (1.f + __expf(-iv));
  fv = 1.f / (1.f + __expf(-fv));
  ov = 1.f / (1.f + __expf(-ov));
  float tg = 2.f / (1.f + __expf(-2.f * gv)) - 1.f;
  float c  = fv * cf[gid] + iv * tg;
  float th = 2.f / (1.f + __expf(-2.f * c)) - 1.f;
  float h  = ov * th;
  cf[gid] = c; hf[gid] = h; hbf[gid] = f2bf(h);
}

// ---------------------------------------------------------------------------
// out[node] = h[node,:] . W_out + b_out   (wave per node, shuffle reduce)
// ---------------------------------------------------------------------------
__global__ __launch_bounds__(256)
void out_proj(const float* __restrict__ hf, const float* __restrict__ Wout,
              const float* __restrict__ bout, float* __restrict__ out)
{
  int wave = threadIdx.x >> 5;
  int lane = threadIdx.x & 31;
  int node = blockIdx.x * 8 + wave;
  if (node >= NNODES) return;
  const float* h = hf + (long)node * HDIM;
  float s = 0.f;
  for (int k = lane; k < HDIM; k += 32) s += h[k] * Wout[k];
  for (int off = 16; off > 0; off >>= 1) s += __shfl_down(s, off, 32);
  if (lane == 0) out[node] = s + bout[0];
}

extern "C" void kernel_launch(void* const* d_in, const int* in_sizes, int n_in,
                              void* d_out, int out_size, void* d_ws, size_t ws_size,
                              hipStream_t stream)
{
  const float* x    = (const float*)d_in[0];
  const float* W_in = (const float*)d_in[1];
  const float* b_in = (const float*)d_in[2];
  const float* Wx   = (const float*)d_in[3];
  const float* Wh   = (const float*)d_in[4];
  const float* Wn   = (const float*)d_in[5];
  const float* b_g  = (const float*)d_in[6];
  const float* Wout = (const float*)d_in[7];
  const float* bout = (const float*)d_in[8];
  const long long* eidx = (const long long*)d_in[9];
  (void)in_sizes; (void)n_in; (void)out_size; (void)ws_size;

  char* wp = (char*)d_ws;
  auto carve = [&](size_t bytes) { char* p = wp; wp += (bytes + 255) & ~size_t(255); return p; };
  u16*   xh    = (u16*)  carve((size_t)T_STEPS * NNODES * HDIM * 2);  // 64 MB
  u16*   WxWnT = (u16*)  carve((size_t)GDIM * 1024 * 2);              // 4 MB
  u16*   WhT   = (u16*)  carve((size_t)GDIM * HDIM * 2);              // 2 MB
  u16*   WinT  = (u16*)  carve((size_t)HDIM * FIN * 2);
  float* pre   = (float*)carve((size_t)NNODES * GDIM * 4);            // 16 MB
  float* gat   = (float*)carve((size_t)NNODES * GDIM * 4);            // 16 MB
  u16*   hbf   = (u16*)  carve((size_t)NNODES * HDIM * 2);
  float* hf    = (float*)carve((size_t)NNODES * HDIM * 4);
  float* cf    = (float*)carve((size_t)NNODES * HDIM * 4);
  u16*   agg   = (u16*)  carve((size_t)NNODES * HDIM * 2);

  // ---- weight prep: transpose + bf16 convert (B stored as Nout x K) ----
  {
    long tot = (long)HDIM * GDIM; int blk = (int)((tot + 255) / 256);
    transpose_cvt<<<blk, 256, 0, stream>>>(WxWnT, Wx, HDIM, GDIM, 1024, 0);
    transpose_cvt<<<blk, 256, 0, stream>>>(WxWnT, Wn, HDIM, GDIM, 1024, 512);
    transpose_cvt<<<blk, 256, 0, stream>>>(WhT,   Wh, HDIM, GDIM, 512,  0);
    long tot2 = (long)FIN * HDIM;
    transpose_cvt<<<(int)((tot2 + 255) / 256), 256, 0, stream>>>(WinT, W_in, FIN, HDIM, FIN, 0);
  }

  // ---- embed: xh = x @ W_in + b_in for all T; t=0 rows init h/c carries ----
  {
    dim3 grid(HDIM / 128, (T_STEPS * NNODES + 127) / 128);
    gemm_wmma<1, 0><<<grid, 256, 0, stream>>>(
        nullptr, nullptr, x, WinT, T_STEPS * NNODES, FIN, HDIM,
        b_in, nullptr, nullptr, xh, hbf, hf, cf, NNODES);
  }

  dim3 gridG(GDIM / 128, (NNODES + 127) / 128);
  for (int t = 0; t < T_STEPS; ++t) {
    // agg = segment_sum(h[src], dst)
    long nq = (long)NNODES * HDIM / 8;
    zero_u16<<<(int)((nq + 255) / 256), 256, 0, stream>>>(agg, nq);
    scatter_pk<<<EDGES * (HDIM / 8) / 256, 256, 0, stream>>>(
        eidx, (long)t * EDGES, (long)T_STEPS * EDGES + (long)t * EDGES, hbf, agg);
    // pre = [xh_t | agg] @ [Wx;Wn] + b_g   (K = 1024)
    gemm_wmma<2, 1><<<gridG, 256, 0, stream>>>(
        xh + (size_t)t * NNODES * HDIM, agg, nullptr, WxWnT,
        NNODES, 1024, GDIM, b_g, nullptr, pre, nullptr, nullptr, nullptr, nullptr, 0);
    for (int l = 0; l < LAYERS; ++l) {
      // gates = pre + h @ Wh
      gemm_wmma<0, 2><<<gridG, 256, 0, stream>>>(
          hbf, nullptr, nullptr, WhT, NNODES, HDIM, GDIM,
          nullptr, pre, gat, nullptr, nullptr, nullptr, nullptr, 0);
      lstm_cell<<<(NNODES * HDIM) / 256, 256, 0, stream>>>(gat, cf, hf, hbf);
    }
  }
  out_proj<<<(NNODES + 7) / 8, 256, 0, stream>>>(hf, Wout, bout, (float*)d_out);
}
